// VarianceLoss_26139170963579
// MI455X (gfx1250) — matrix operand order repaired and checked by reading, not compiled, not run on hardware
//
#include <hip/hip_runtime.h>
#include <hip/hip_bf16.h>

typedef __attribute__((ext_vector_type(16))) _Float16 v16h;
typedef __attribute__((ext_vector_type(8)))  float    v8f;

#define BB     8
#define NN     200
#define KK     32
#define HH     138
#define WWD    138
#define HWPIX  (HH * WWD)          // 19044
#define HIMG   550
#define WIMG   550
#define EPS    1e-6f
#define PT     ((HWPIX + 15) / 16) // 1191 pixel tiles per batch image

__device__ __forceinline__ float fast_rcp(float x) {
  return __builtin_amdgcn_rcpf(x);           // v_rcp_f32 (TRANS, co-executes)
}

// ---------------------------------------------------------------------------
// Pass 1: fused GEMM (proto x mask^T) + sigmoid + gaussian + conf weighting +
// reduction over priors N.  One wave per 16-pixel tile, WMMA 16x16x32 f16.
//   final_conf[b,p] = 1 - (sum_n m^2) / (sum_n m + eps),
//   m = sigmoid(proto.mask) * gauss(loc) * conf
// ---------------------------------------------------------------------------
__global__ __launch_bounds__(256) void conf_gemm_kernel(
    const float* __restrict__ loc,    // [B,N,4]
    const float* __restrict__ conf,   // [B,N]
    const float* __restrict__ mask,   // [B,N,K]
    const float* __restrict__ proto,  // [B,H,W,K]
    float* __restrict__ final_conf)   // [B,H*W]
{
  const int wave = (blockIdx.x * blockDim.x + threadIdx.x) >> 5;
  const int lane = threadIdx.x & 31;
  if (wave >= BB * PT) return;              // wave-uniform (exact grid anyway)
  const int b    = wave / PT;
  const int pt   = wave % PT;
  const int p0   = pt * 16;
  const int half = lane >> 4;               // 0 = lanes 0-15, 1 = lanes 16-31
  const int l16  = lane & 15;

  // ---- A tile: 16 pixels x 32 K, f16, ISA 16-bit A layout ----
  // half=0 row M=l16 holds K {0..7, 16..23}; half=1 holds K {8..15, 24..31}
  const int pixA = p0 + l16;
  const int kb0  = half ? 8 : 0;            // multiple of 4 -> float4-aligned
  v16h a;
  if (pixA < HWPIX) {
    const float4* ap4 =
        reinterpret_cast<const float4*>(proto + ((size_t)b * HWPIX + pixA) * KK);
    float4 f0 = ap4[(kb0 >> 2) + 0];        // K kb0 .. kb0+3
    float4 f1 = ap4[(kb0 >> 2) + 1];        // K kb0+4 .. kb0+7
    float4 f2 = ap4[(kb0 >> 2) + 4];        // K kb0+16 .. kb0+19
    float4 f3 = ap4[(kb0 >> 2) + 5];        // K kb0+20 .. kb0+23
    a[0]=(_Float16)f0.x; a[1]=(_Float16)f0.y; a[2]=(_Float16)f0.z; a[3]=(_Float16)f0.w;
    a[4]=(_Float16)f1.x; a[5]=(_Float16)f1.y; a[6]=(_Float16)f1.z; a[7]=(_Float16)f1.w;
    a[8]=(_Float16)f2.x; a[9]=(_Float16)f2.y; a[10]=(_Float16)f2.z; a[11]=(_Float16)f2.w;
    a[12]=(_Float16)f3.x; a[13]=(_Float16)f3.y; a[14]=(_Float16)f3.z; a[15]=(_Float16)f3.w;
  } else {
#pragma unroll
    for (int i = 0; i < 16; ++i) a[i] = (_Float16)0.0f;
  }

  // Normalized coords of the 8 C-rows this lane owns (row M = r + 8*half)
  float xs[8], ys[8];
  int   pixr[8];
#pragma unroll
  for (int r = 0; r < 8; ++r) {
    int p   = p0 + r + (half ? 8 : 0);
    pixr[r] = p;
    int pc  = (p < HWPIX) ? p : 0;
    int h   = pc / WWD, w = pc % WWD;
    xs[r] = ((float)w + 0.5f) * (1.0f / (float)WWD);
    ys[r] = ((float)h + 0.5f) * (1.0f / (float)HH);
  }

  float sm[8], sm2[8];
#pragma unroll
  for (int r = 0; r < 8; ++r) { sm[r] = 0.0f; sm2[r] = 0.0f; }

  const int NT = (NN + 15) / 16;            // 13 prior tiles
  for (int nt = 0; nt < NT; ++nt) {
    const int n = nt * 16 + l16;            // this lane's prior column
    float cx, cy, isx, isy, cf;
    if (n < NN) {
      const float4 lp = *reinterpret_cast<const float4*>(loc + ((size_t)b * NN + n) * 4);
      cx = lp.x; cy = lp.y;
      isx = fast_rcp(lp.z);
      isy = fast_rcp(lp.w);
      cf = conf[(size_t)b * NN + n];
    } else { cx = 0.f; cy = 0.f; isx = 1.f; isy = 1.f; cf = 0.f; }

    // ---- B tile: 32 K x 16 priors, ISA 16-bit B layout ----
    // lane holds column n; half=0 -> K {0..7, 16..23}, half=1 -> K {8..15, 24..31}
    v16h bv;
    if (n < NN) {
      const float4* bp4 = reinterpret_cast<const float4*>(
          mask + ((size_t)b * NN + n) * KK + (half ? 8 : 0));
#pragma unroll
      for (int q = 0; q < 4; ++q) {
        float4 f = bp4[(q & 1) + ((q >> 1) * 4)];
        bv[4*q+0]=(_Float16)f.x; bv[4*q+1]=(_Float16)f.y;
        bv[4*q+2]=(_Float16)f.z; bv[4*q+3]=(_Float16)f.w;
      }
    } else {
#pragma unroll
      for (int i = 0; i < 16; ++i) bv[i] = (_Float16)0.0f;
    }

    v8f c = {};
    c = __builtin_amdgcn_wmma_f32_16x16x32_f16(
        /*neg_a=*/false, a, /*neg_b=*/false, bv,
        /*c_mod=*/(short)0, c, /*reuse_a=*/false, /*reuse_b=*/false);

#pragma unroll
    for (int r = 0; r < 8; ++r) {
      float s  = fast_rcp(1.0f + __expf(-c[r]));    // sigmoid(logit)
      float dx = (xs[r] - cx) * isx;
      float dy = (ys[r] - cy) * isy;
      float g  = __expf(-0.5f * (dx * dx + dy * dy));
      float m  = s * g * cf;
      sm[r]  += m;
      sm2[r] += m * m;
    }
  }

  // Reduce over the 16 prior columns held by this half-wave
#pragma unroll
  for (int r = 0; r < 8; ++r) {
    float v = sm[r], v2 = sm2[r];
#pragma unroll
    for (int off = 1; off < 16; off <<= 1) {
      v  += __shfl_xor(v,  off, 16);
      v2 += __shfl_xor(v2, off, 16);
    }
    sm[r] = v; sm2[r] = v2;
  }

  if (l16 == 0) {
#pragma unroll
    for (int r = 0; r < 8; ++r) {
      int p = pixr[r];
      if (p < HWPIX)
        final_conf[(size_t)b * HWPIX + p] = 1.0f - sm2[r] * fast_rcp(sm[r] + EPS);
    }
  }
}

// ---------------------------------------------------------------------------
// Pass 2: fused bilinear upsample (recomputed taps, final_conf is L2-resident)
// + total_conf + weighted_mean + weighted variance, per 550x550 pixel.
// Deterministic block partial sums -> ws.
// ---------------------------------------------------------------------------
__global__ __launch_bounds__(256) void variance_kernel(
    const float* __restrict__ original,    // [B,3,HIMG,WIMG]
    const float* __restrict__ final_conf,  // [B,H*W]
    float* __restrict__ partials)
{
  __shared__ float red[256];
  const int idx = blockIdx.x * blockDim.x + threadIdx.x;
  float acc = 0.0f;
  if (idx < HIMG * WIMG) {
    const int h = idx / WIMG;
    const int w = idx % WIMG;

    // half-pixel-center bilinear source coords (align_corners=False)
    float sy = ((float)h + 0.5f) * ((float)HH  / (float)HIMG) - 0.5f;
    float sx = ((float)w + 0.5f) * ((float)WWD / (float)WIMG) - 0.5f;
    float y0f = floorf(sy), x0f = floorf(sx);
    float wy = sy - y0f,    wx = sx - x0f;
    int y0 = (int)y0f, x0 = (int)x0f;
    int y1 = y0 + 1,   x1 = x0 + 1;
    y0 = min(max(y0, 0), HH - 1);  y1 = min(max(y1, 0), HH - 1);
    x0 = min(max(x0, 0), WWD - 1); x1 = min(max(x1, 0), WWD - 1);
    const float w00 = (1.f - wy) * (1.f - wx), w01 = (1.f - wy) * wx;
    const float w10 = wy * (1.f - wx),         w11 = wy * wx;

    float r[BB];
    float total = 0.0f;
#pragma unroll
    for (int b = 0; b < BB; ++b) {
      const float* f = final_conf + (size_t)b * HWPIX;
      float v = w00 * f[y0 * WWD + x0] + w01 * f[y0 * WWD + x1]
              + w10 * f[y1 * WWD + x0] + w11 * f[y1 * WWD + x1];
      r[b] = v;
      total += v;
    }
    const float inv = fast_rcp(total + EPS);

#pragma unroll
    for (int c = 0; c < 3; ++c) {
      float o[BB];
      float wm = 0.0f;                      // weighted "mean" (unnormalized sum)
#pragma unroll
      for (int b = 0; b < BB; ++b) {
        const float* op = original + (((size_t)b * 3 + c) * HIMG + h) * WIMG + w;
        __builtin_prefetch(op + WIMG, 0, 3);   // global_prefetch_b8 (next row)
        o[b] = *op;
        wm  += o[b] * r[b];
      }
#pragma unroll
      for (int b = 0; b < BB; ++b) {
        float d = o[b] - wm;
        acc += d * d * r[b];
      }
    }
    acc *= inv;
  }

  red[threadIdx.x] = acc;
  __syncthreads();
  for (int s = 128; s > 0; s >>= 1) {
    if ((int)threadIdx.x < s) red[threadIdx.x] += red[threadIdx.x + s];
    __syncthreads();
  }
  if (threadIdx.x == 0) partials[blockIdx.x] = red[0];
}

// Final deterministic reduction of per-block partials -> scalar output
__global__ __launch_bounds__(256) void reduce_kernel(
    const float* __restrict__ partials, int n, float* __restrict__ out)
{
  __shared__ float red[256];
  float acc = 0.0f;
  for (int i = threadIdx.x; i < n; i += 256) acc += partials[i];
  red[threadIdx.x] = acc;
  __syncthreads();
  for (int s = 128; s > 0; s >>= 1) {
    if ((int)threadIdx.x < s) red[threadIdx.x] += red[threadIdx.x + s];
    __syncthreads();
  }
  if (threadIdx.x == 0) out[0] = red[0];
}

extern "C" void kernel_launch(void* const* d_in, const int* in_sizes, int n_in,
                              void* d_out, int out_size, void* d_ws, size_t ws_size,
                              hipStream_t stream) {
  (void)in_sizes; (void)n_in; (void)out_size; (void)ws_size;
  const float* original = (const float*)d_in[0];  // [8,3,550,550]
  const float* loc      = (const float*)d_in[1];  // [8,200,4]
  const float* conf     = (const float*)d_in[2];  // [8,200]
  const float* mask     = (const float*)d_in[3];  // [8,200,32]
  const float* proto    = (const float*)d_in[4];  // [8,138,138,32]
  float* out = (float*)d_out;

  float* fc       = (float*)d_ws;                 // [8 * 19044] final_conf
  float* partials = fc + (size_t)BB * HWPIX;      // per-block partials

  // Pass 1: 8 waves/block, one wave per 16-pixel tile -> exact grid
  const int total_waves = BB * PT;                // 9528
  const int gemm_blocks = total_waves / 8;        // 1191 (divides exactly)
  conf_gemm_kernel<<<gemm_blocks, 256, 0, stream>>>(loc, conf, mask, proto, fc);

  // Pass 2: one thread per output pixel
  const int pix     = HIMG * WIMG;                // 302500
  const int vblocks = (pix + 255) / 256;          // 1182
  variance_kernel<<<vblocks, 256, 0, stream>>>(original, fc, partials);

  reduce_kernel<<<1, 256, 0, stream>>>(partials, vblocks, out);
}
